// LocalAGLKABranch_21784074125697
// MI455X (gfx1250) — matrix operand order repaired and checked by reading, not compiled, not run on hardware
//
#include <hip/hip_runtime.h>
#include <hip/hip_bf16.h>
#include <math.h>

// ---------------- problem constants ----------------
#define B_   32
#define C_   256
#define H_   56
#define W_   56
#define HW_  (H_ * W_)            // 3136
#define NEL  (B_ * C_ * HW_)      // 25,690,112
#define NPIX (B_ * HW_)           // 100,352
#define GN_G 8
#define EPS_ 1e-5f

// ---------------- WMMA / async types ----------------
typedef __attribute__((ext_vector_type(16))) __bf16 v16bf;
typedef __attribute__((ext_vector_type(8)))  float  v8f;
typedef int vint4 __attribute__((vector_size(4 * sizeof(int))));

union BfFrag { v16bf v; unsigned int u[8]; };

#if defined(__has_builtin)
#if __has_builtin(__builtin_amdgcn_global_load_async_to_lds_b128)
#define HAVE_ASYNC_LDS 1
#endif
#endif
#ifndef HAVE_ASYNC_LDS
#define HAVE_ASYNC_LDS 0
#endif

#define AS1 __attribute__((address_space(1)))
#define AS3 __attribute__((address_space(3)))

__device__ __forceinline__ unsigned short f2bf(float f) {
    unsigned int b = __float_as_uint(f);
    b += 0x7FFFu + ((b >> 16) & 1u);   // round-to-nearest-even
    return (unsigned short)(b >> 16);
}
__device__ __forceinline__ float silu_f(float x) { return x / (1.f + __expf(-x)); }

// ---------------- K0: zero GN stats ----------------
__global__ void k_zero_stats(float* stats) { stats[threadIdx.x] = 0.f; } // <<<1,512>>>

// ---------------- K1: convert w_mix -> bf16 ----------------
__global__ void k_cvt_wmix(const float* __restrict__ w, unsigned short* __restrict__ wb) {
    int i = blockIdx.x * 256 + threadIdx.x;        // 65536 total
    wb[i] = f2bf(w[i]);
}

// ---------------- K2: channel LayerNorm ----------------
// one thread per pixel (b,h,w); loops over C (coalesced across w)
__global__ void k_ln(const float* __restrict__ x, const float* __restrict__ lng,
                     const float* __restrict__ lnb, float* __restrict__ x1) {
    int p  = blockIdx.x * 256 + threadIdx.x;       // 0..NPIX-1 (exact)
    int b  = p / HW_;
    int hw = p - b * HW_;
    const float* xp = x + (size_t)b * C_ * HW_ + hw;
    float sum = 0.f, sq = 0.f;
    for (int c = 0; c < C_; ++c) {
        float v = xp[(size_t)c * HW_];
        sum += v; sq += v * v;
    }
    float mean = sum * (1.f / C_);
    float var  = sq * (1.f / C_) - mean * mean;
    float inv  = rsqrtf(var + EPS_);
    float* op = x1 + (size_t)b * C_ * HW_ + hw;
    for (int c = 0; c < C_; ++c) {
        float v = xp[(size_t)c * HW_];
        op[(size_t)c * HW_] = (v - mean) * inv * lng[c] + lnb[c];
    }
}

// ---------------- K3: DW3x3 + SiLU + fused GAP ----------------
// one block per (b,c) plane -> single (non-atomic) store of the plane mean
__global__ void k_dw3_gap(const float* __restrict__ x1, const float* __restrict__ w3,
                          float* __restrict__ x2, float* __restrict__ gap) {
    int bc = blockIdx.x;                 // 0..B*C-1
    int c  = bc & (C_ - 1);
    const float* xp = x1 + (size_t)bc * HW_;
    float* op = x2 + (size_t)bc * HW_;
    __shared__ float wsm[9];
    __shared__ float red[256];
    int t = threadIdx.x;
    if (t < 9) wsm[t] = w3[c * 9 + t];
    __syncthreads();
    float lsum = 0.f;
    for (int p = t; p < HW_; p += 256) {
        int h = p / W_, w = p - h * W_;
        float acc = 0.f;
#pragma unroll
        for (int kh = 0; kh < 3; ++kh) {
            int hh = h + kh - 1;
            if (hh < 0 || hh >= H_) continue;
#pragma unroll
            for (int kw = 0; kw < 3; ++kw) {
                int ww = w + kw - 1;
                if (ww < 0 || ww >= W_) continue;
                acc += wsm[kh * 3 + kw] * xp[hh * W_ + ww];
            }
        }
        float sv = silu_f(acc);
        op[p] = sv;
        lsum += sv;
    }
    red[t] = lsum;
    __syncthreads();
    for (int s = 128; s; s >>= 1) {
        if (t < s) red[t] += red[t + s];
        __syncthreads();
    }
    if (t == 0) gap[bc] = red[0] * (1.f / HW_);
}

// ---------------- K4: spatial gate ----------------
__global__ void k_gate(const float* __restrict__ gap, const float* __restrict__ wg,
                       float* __restrict__ gv) {
    int b = blockIdx.x, o = threadIdx.x;
    __shared__ float gs[C_];
    gs[o] = gap[b * C_ + o];
    __syncthreads();
    float acc = 0.f;
    for (int c = 0; c < C_; ++c) acc += gs[c] * wg[o * C_ + c];
    gv[b * C_ + o] = 1.f / (1.f + __expf(-acc));
}

// ---------------- K5: multi-branch large-kernel DW convs + gate mul ----------------
#define CHUNKS 13   // ceil(3136/256)
__global__ void k_branches(const float* __restrict__ x2, const float* __restrict__ w5,
                           const float* __restrict__ w7, const float* __restrict__ w19,
                           const float* __restrict__ w71, const float* __restrict__ gv,
                           float* __restrict__ s) {
    int bid   = blockIdx.x;
    int plane = bid / CHUNKS;
    int chunk = bid - plane * CHUNKS;
    int c = plane & (C_ - 1);
    __shared__ float W5[25], W7[49], W19[9], W71[7];
    int t = threadIdx.x;
    if (t < 25)                   W5[t]        = w5 [c * 25 + t];
    else if (t >= 32  && t < 81)  W7[t - 32]   = w7 [c * 49 + (t - 32)];
    else if (t >= 96  && t < 105) W19[t - 96]  = w19[c * 9  + (t - 96)];
    else if (t >= 112 && t < 119) W71[t - 112] = w71[c * 7  + (t - 112)];
    __syncthreads();
    int p = chunk * 256 + t;
    if (p >= HW_) return;
    const float* xp = x2 + (size_t)plane * HW_;
    int h = p / W_, w = p - h * W_;
    float acc = 0.f;
    // DW 5x5, pad 2
#pragma unroll
    for (int kh = 0; kh < 5; ++kh) {
        int hh = h + kh - 2;
        if (hh < 0 || hh >= H_) continue;
#pragma unroll
        for (int kw = 0; kw < 5; ++kw) {
            int ww = w + kw - 2;
            if (ww < 0 || ww >= W_) continue;
            acc += W5[kh * 5 + kw] * xp[hh * W_ + ww];
        }
    }
    // DW 7x7, dilation 2, pad 6
#pragma unroll
    for (int kh = 0; kh < 7; ++kh) {
        int hh = h + (kh - 3) * 2;
        if (hh < 0 || hh >= H_) continue;
#pragma unroll
        for (int kw = 0; kw < 7; ++kw) {
            int ww = w + (kw - 3) * 2;
            if (ww < 0 || ww >= W_) continue;
            acc += W7[kh * 7 + kw] * xp[hh * W_ + ww];
        }
    }
    // DW 1x9, pad (0,4)
#pragma unroll
    for (int kw = 0; kw < 9; ++kw) {
        int ww = w + kw - 4;
        if (ww < 0 || ww >= W_) continue;
        acc += W19[kw] * xp[h * W_ + ww];
    }
    // DW 7x1, pad (3,0)
#pragma unroll
    for (int kh = 0; kh < 7; ++kh) {
        int hh = h + kh - 3;
        if (hh < 0 || hh >= H_) continue;
        acc += W71[kh] * xp[hh * W_ + w];
    }
    s[(size_t)plane * HW_ + p] = acc * gv[plane];
}

// ---------------- K6: 1x1 channel-mix GEMM (WMMA bf16) + fused GN stats ----------------
// per batch b: M[256 o] x K[256 c] x N[3136 p].  Block = 256 thr = 8 waves.
// Block tile 128M x 32N, K-step 32. Wave w -> M-subtile w (16 rows), 2 N-subtiles.
#define BM 128
#define BN 32
#define BK 32
__global__ void k_mix_gemm_gn(const float* __restrict__ s,
                              const unsigned short* __restrict__ wb, // bf16 w_mix [256][256]
                              float* __restrict__ m,
                              float* __restrict__ stats) {
    __shared__ unsigned short As[BM * BK];   // [o_local][c_local] (pairs contiguous in c)
    __shared__ unsigned short Bs[BN * BK];   // [p_local][c_local] (pairs contiguous in c)
    int b  = blockIdx.z;
    int Mo = blockIdx.y * BM;
    int Np = blockIdx.x * BN;
    int t    = threadIdx.x;
    int wave = t >> 5;
    int lane = t & 31;
    v8f acc0 = {}, acc1 = {};
    const float* sB = s + (size_t)b * C_ * HW_;
    const unsigned int* wsrc = (const unsigned int*)wb;   // [256][128] packed bf16 pairs

    for (int k0 = 0; k0 < C_; k0 += BK) {
        __syncthreads();
        // ---- stage A: 128x32 bf16 tile (8 KB, already bf16 in global) ----
        // 512 x 16B transfers; async direct-to-LDS when available (ASYNCcnt path)
#pragma unroll
        for (int r = 0; r < 2; ++r) {
            int i = t + r * 256;                 // 16B-chunk index 0..511
            int o = i >> 2, q = i & 3;           // row, 16B-chunk within 64B row slab
            const unsigned int* g = wsrc + (size_t)(Mo + o) * (C_ / 2) + (k0 >> 1) + q * 4;
            unsigned int* l = (unsigned int*)As + i * 4;
#if HAVE_ASYNC_LDS
            __builtin_amdgcn_global_load_async_to_lds_b128(
                (AS1 vint4*)g, (AS3 vint4*)l, 0, 0);
#else
            *(uint4*)l = *(const uint4*)g;
#endif
        }
        // ---- stage B: 32c x 32p f32 -> packed bf16 pairs, transposed to [p][c] ----
        // 512 c-pairs, 2 per thread, one ds_store_b32 each
#pragma unroll
        for (int r = 0; r < 2; ++r) {
            int i  = t + r * 256;                // 0..511
            int cp = i >> 5;                     // c-pair 0..15
            int pl = i & 31;
            float v0 = sB[(size_t)(k0 + 2 * cp)     * HW_ + Np + pl];
            float v1 = sB[(size_t)(k0 + 2 * cp + 1) * HW_ + Np + pl];
            unsigned int pk = (unsigned int)f2bf(v0) | ((unsigned int)f2bf(v1) << 16);
            ((unsigned int*)Bs)[pl * 16 + cp] = pk;
        }
        if (k0 + BK < C_)   // prefetch next K-slab of s -> global_prefetch_b8
            __builtin_prefetch(&sB[(size_t)(k0 + BK) * HW_ + Np + (t & 31)], 0, 0);
#if HAVE_ASYNC_LDS
        asm volatile("s_wait_asynccnt 0" ::: "memory");
#endif
        __syncthreads();

        // A fragment (ISA 16-bit A 16x32 layout): lane<16 -> K pairs {0..3,8..11}, else {4..7,12..15}
        int mrow = wave * 16 + (lane & 15);
        int kbA  = (lane < 16) ? 0 : 4;
        const unsigned int* Au = (const unsigned int*)As;
        BfFrag a;
#pragma unroll
        for (int i2 = 0; i2 < 4; ++i2) {
            a.u[i2]     = Au[mrow * 16 + kbA + i2];
            a.u[4 + i2] = Au[mrow * 16 + kbA + 8 + i2];
        }
        // B fragments: lane<16 -> K pairs {0..3,8..11}, else {4..7,12..15}
        int nl  = lane & 15;
        int kbB = (lane < 16) ? 0 : 4;
        const unsigned int* Bu = (const unsigned int*)Bs;
        BfFrag b0, b1;
#pragma unroll
        for (int i2 = 0; i2 < 4; ++i2) {
            b0.u[i2]     = Bu[nl * 16 + kbB + i2];
            b0.u[4 + i2] = Bu[nl * 16 + kbB + 8 + i2];
            b1.u[i2]     = Bu[(nl + 16) * 16 + kbB + i2];
            b1.u[4 + i2] = Bu[(nl + 16) * 16 + kbB + 8 + i2];
        }
        acc0 = __builtin_amdgcn_wmma_f32_16x16x32_bf16(false, a.v, false, b0.v,
                                                       (short)0, acc0, false, false);
        acc1 = __builtin_amdgcn_wmma_f32_16x16x32_bf16(false, a.v, false, b1.v,
                                                       (short)0, acc1, false, false);
    }

    // epilogue: write m, accumulate GroupNorm sum/sumsq (16-row tile lies in one group)
    int mbase = Mo + wave * 16 + ((lane < 16) ? 0 : 8);
    int n0 = Np + (lane & 15), n1 = n0 + 16;
    float* mB = m + (size_t)b * C_ * HW_;
    float lsum = 0.f, lsq = 0.f;
#pragma unroll
    for (int r = 0; r < 8; ++r) {
        float v0 = acc0[r], v1 = acc1[r];
        mB[(size_t)(mbase + r) * HW_ + n0] = v0;
        mB[(size_t)(mbase + r) * HW_ + n1] = v1;
        lsum += v0 + v1;
        lsq  += v0 * v0 + v1 * v1;
    }
    for (int off = 16; off; off >>= 1) {
        lsum += __shfl_xor(lsum, off, 32);
        lsq  += __shfl_xor(lsq,  off, 32);
    }
    if (lane == 0) {
        int g = (Mo + wave * 16) >> 5;     // 32 channels per group
        atomicAdd(&stats[b * GN_G + g], lsum);
        atomicAdd(&stats[B_ * GN_G + b * GN_G + g], lsq);
    }
}

// ---------------- K7: GN normalize + SiLU + residuals ----------------
__global__ void k_finalize(const float* __restrict__ x, const float* __restrict__ s,
                           const float* __restrict__ m, const float* __restrict__ stats,
                           const float* __restrict__ gng, const float* __restrict__ gnb,
                           const float* __restrict__ alpha_p, float* __restrict__ out) {
    size_t idx = (size_t)blockIdx.x * 256 + threadIdx.x;   // exact: NEL = 100352*256
    int b = (int)(idx / ((size_t)C_ * HW_));
    int c = (int)((idx / HW_) & (C_ - 1));
    int g = c >> 5;
    const float inv_n = 1.f / ((C_ / GN_G) * HW_);
    float mean = stats[b * GN_G + g] * inv_n;
    float var  = stats[B_ * GN_G + b * GN_G + g] * inv_n - mean * mean;
    float inv  = rsqrtf(var + EPS_);
    float mv   = (m[idx] - mean) * inv * gng[c] + gnb[c];
    float f    = s[idx] + alpha_p[0] * silu_f(mv);
    out[idx]   = x[idx] + 0.1f * f;
}

// ---------------- launcher ----------------
extern "C" void kernel_launch(void* const* d_in, const int* in_sizes, int n_in,
                              void* d_out, int out_size, void* d_ws, size_t ws_size,
                              hipStream_t stream) {
    const float* x     = (const float*)d_in[0];
    const float* ln_g  = (const float*)d_in[1];
    const float* ln_b  = (const float*)d_in[2];
    const float* w_dw3 = (const float*)d_in[3];
    const float* w_dw5 = (const float*)d_in[4];
    const float* w_dw7 = (const float*)d_in[5];
    const float* w_1x9 = (const float*)d_in[6];
    const float* w_7x1 = (const float*)d_in[7];
    const float* w_gate= (const float*)d_in[8];
    const float* w_mix = (const float*)d_in[9];
    const float* gn_g  = (const float*)d_in[10];
    const float* gn_b  = (const float*)d_in[11];
    const float* alpha = (const float*)d_in[12];
    float* out = (float*)d_out;

    // workspace layout (all offsets 256B aligned)
    const size_t S = (size_t)NEL * sizeof(float);   // 102,760,448
    char* ws = (char*)d_ws;
    float*          x2    = (float*)(ws);               // DW3 output
    float*          x1m   = (float*)(ws + S);           // x1 (LN out), reused as m (mix out)
    float*          sbuf  = (float*)(ws + 2 * S);       // branch sum * gate
    float*          gap   = (float*)(ws + 3 * S);                     // B*C
    float*          gv    = (float*)(ws + 3 * S + 32768);             // B*C
    float*          stats = (float*)(ws + 3 * S + 65536);             // 512 floats
    unsigned short* wmixb = (unsigned short*)(ws + 3 * S + 65536 + 2048);

    k_zero_stats<<<1, 512, 0, stream>>>(stats);
    k_cvt_wmix<<<(C_ * C_) / 256, 256, 0, stream>>>(w_mix, wmixb);
    k_ln<<<NPIX / 256, 256, 0, stream>>>(x, ln_g, ln_b, x1m);
    k_dw3_gap<<<B_ * C_, 256, 0, stream>>>(x1m, w_dw3, x2, gap);
    k_gate<<<B_, C_, 0, stream>>>(gap, w_gate, gv);
    k_branches<<<B_ * C_ * CHUNKS, 256, 0, stream>>>(x2, w_dw5, w_dw7, w_1x9, w_7x1, gv, sbuf);
    // m overwrites x1 slot (x1 dead after k_dw3_gap)
    k_mix_gemm_gn<<<dim3(HW_ / BN, C_ / BM, B_), 256, 0, stream>>>(sbuf, wmixb, x1m, stats);
    k_finalize<<<NEL / 256, 256, 0, stream>>>(x, sbuf, x1m, stats, gn_g, gn_b, alpha, out);
}
